// hypergraph_constructor_17300128268697
// MI455X (gfx1250) — matrix operand order repaired and checked by reading, not compiled, not run on hardware
//
#include <hip/hip_runtime.h>
#include <hip/hip_bf16.h>
#include <math.h>

// ---------------------------------------------------------------------------
// Hypergraph constructor for MI455X (gfx1250, wave32, WMMA + TDM).
//   nv1 = tanh(3*(embn@W1+b1))        [10000,128] -> f16 (ws)
//   nv2 = tanh(3*(embhe@W2+b2))       [  512,128] -> f16 (ws)
//   H   = relu(tanh(3*(nv1@nv2^T)))   [10000,512] -> f16 (ws)      (WMMA)
//   adj = H@H^T                       [10000,10000] -> f32 (d_out) (WMMA, 102 GF)
//   per-row exact top-32 radix select + index-order-tie mask, in place.
// A/B panels are staged into LDS by the Tensor Data Mover (double buffered,
// TENSORcnt-tracked), not by per-thread load/ds_store round trips.
// ---------------------------------------------------------------------------

typedef __attribute__((ext_vector_type(16))) _Float16 v16h;
typedef __attribute__((ext_vector_type(8)))  _Float16 v8h;
typedef __attribute__((ext_vector_type(8)))  float    v8f;

typedef unsigned int uint32x4 __attribute__((ext_vector_type(4)));
typedef int          int32x8  __attribute__((ext_vector_type(8)));
typedef int          int32x4  __attribute__((ext_vector_type(4)));

#define NNODES  10000
#define NHEDGES 512
#define DIM     128
#define KSEL    32
#define ALPHA   3.0f

// --------------------------- stage 1/2: embed MLP ---------------------------
__global__ __launch_bounds__(128) void embed_mlp_kernel(
    const float* __restrict__ E, const float* __restrict__ W,
    const float* __restrict__ b, const int* __restrict__ idx,
    _Float16* __restrict__ out, int nrows)
{
    __shared__ float Es[DIM];
    const int row = blockIdx.x;
    const int c   = threadIdx.x;
    const int src = idx ? idx[row] : row;
    Es[c] = E[(size_t)src * DIM + c];
    __syncthreads();
    float acc = b[c];
#pragma unroll 8
    for (int k = 0; k < DIM; ++k)
        acc = fmaf(Es[k], W[(size_t)k * DIM + c], acc);
    out[(size_t)row * DIM + c] = (_Float16)tanhf(ALPHA * acc);
}

// -------------------- TDM: 128-row x 32-half 2D tile -> LDS -----------------
// D# per CDNA5 ISA ch.8: group0 {count, lds_addr, global_addr, type=2},
// group1 {data_size=2B, pad 4 DWORDs every 16 DWORDs (64B rows -> 80B pitch),
//         tensor_dim0/1 = remaining K/rows (OOB reads return 0 for edge tiles),
//         tile = 32 x 128, tensor_dim0_stride = ld}.
__device__ __forceinline__ void tdm_load_panel(
    const _Float16* gsrc, unsigned lds_off,
    int rows_remaining, int k_remaining, int ld_elems)
{
    const unsigned long long ga = (unsigned long long)(uintptr_t)gsrc;
    uint32x4 g0;
    g0.x = 1u;                                                   // count=1, valid
    g0.y = lds_off;                                              // lds_addr (bytes)
    g0.z = (unsigned)(ga & 0xFFFFFFFFull);                       // global_addr[31:0]
    g0.w = (unsigned)((ga >> 32) & 0x01FFFFFFull) | (2u << 30);  // addr[56:32]|type=2

    int32x8 g1;
    g1[0] = (int)((1u << 16)        // data_size = 1 -> 2 bytes
                | (1u << 20)        // pad_enable
                | (3u << 22)        // pad_interval: every 16 DWORDs (64B)
                | (3u << 25));      // pad_amount:   4 DWORDs (16B)
    g1[1] = (int)(((unsigned)k_remaining & 0xFFFFu) << 16);          // dim0 lo
    g1[2] = (int)((((unsigned)k_remaining >> 16) & 0xFFFFu)
                | (((unsigned)rows_remaining & 0xFFFFu) << 16));     // dim0 hi|dim1 lo
    g1[3] = (int)((((unsigned)rows_remaining >> 16) & 0xFFFFu)
                | (32u << 16));                                      // dim1 hi|tile_dim0
    g1[4] = (int)128u;                                               // tile_dim1 (dim2=0)
    g1[5] = (int)(unsigned)ld_elems;                                 // dim0_stride lo
    g1[6] = 0;                                                       // stride hi
    g1[7] = 0;
    const int32x4 z4 = {0, 0, 0, 0};
#if defined(__clang_major__) && (__clang_major__ >= 23)
    const int32x8 z8 = {0, 0, 0, 0, 0, 0, 0, 0};
    __builtin_amdgcn_tensor_load_to_lds(g0, g1, z4, z4, z8, 0);
#else
    __builtin_amdgcn_tensor_load_to_lds(g0, g1, z4, z4, 0);
#endif
}

// ------------------------ WMMA GEMM-NT: C = A * B^T -------------------------
// A: [M,Kd] f16 row-major, B: [N,Kd] f16 row-major.
// EPI==0: C f32 (adj).  EPI==1: C f16 = relu(tanh(3*x)) (H).
// Block: 128x128 tile, 256 threads = 8 waves; wave (wr,wc) owns 32x64.
#define LDSK 40   // 32 K-halfs + 8 pad halfs (80B pitch, written by TDM padding)

template <int EPI>
__global__ __launch_bounds__(256) void gemm_nt_wmma(
    const _Float16* __restrict__ A, const _Float16* __restrict__ B,
    void* __restrict__ C, int M, int N, int Kd, int ldc)
{
    __shared__ __align__(16) _Float16 As[2][128 * LDSK];
    __shared__ __align__(16) _Float16 Bs[2][128 * LDSK];

    const int tid  = threadIdx.x;
    const int lane = tid & 31;
    const int wave = tid >> 5;
    const int wr   = wave >> 1;          // 0..3 -> wave row (32 rows)
    const int wc   = wave & 1;           // 0..1 -> wave col (64 cols)
    const int lh   = lane & 15;
    const int hi   = lane >> 4;          // lane half

    const int i0 = blockIdx.y * 128;
    const int j0 = blockIdx.x * 128;

    const unsigned asOff    = (unsigned)(uintptr_t)&As[0][0];  // flat low 32 = LDS offset
    const unsigned bsOff    = (unsigned)(uintptr_t)&Bs[0][0];
    const unsigned bufBytes = 128u * LDSK * (unsigned)sizeof(_Float16);

    // one wave drives the Tensor Data Mover; TENSORcnt is per-wave
    if (tid < 32) {
        tdm_load_panel(A + (size_t)i0 * Kd, asOff, M - i0, Kd, Kd);
        tdm_load_panel(B + (size_t)j0 * Kd, bsOff, N - j0, Kd, Kd);
        __builtin_amdgcn_s_wait_tensorcnt(0);
    }
    __syncthreads();

    v8f acc[2][4] = {};
    int cur = 0;

    for (int kk = 0; kk < Kd; kk += 32) {
        // kick off next K-panel into the other buffer while we compute
        if ((kk + 32 < Kd) && tid < 32) {
            const unsigned nb = (unsigned)(cur ^ 1) * bufBytes;
            tdm_load_panel(A + (size_t)i0 * Kd + kk + 32, asOff + nb, M - i0, Kd - kk - 32, Kd);
            tdm_load_panel(B + (size_t)j0 * Kd + kk + 32, bsOff + nb, N - j0, Kd - kk - 32, Kd);
        }

        // ---- build fragments per ISA 16-bit layouts ----
        // A 16x32: lanes 0-15: M=lh, K{0..7,16..23}; lanes 16-31: M=lh, K{8..15,24..31}
        v16h afrag[2];
#pragma unroll
        for (int tm = 0; tm < 2; ++tm) {
            const int ar = wr * 32 + tm * 16 + lh;
            v8h c0 = *reinterpret_cast<const v8h*>(&As[cur][ar * LDSK + (hi ? 8  : 0)]);
            v8h c1 = *reinterpret_cast<const v8h*>(&As[cur][ar * LDSK + (hi ? 24 : 16)]);
            afrag[tm] = __builtin_shufflevector(c0, c1,
                0,1,2,3,4,5,6,7,8,9,10,11,12,13,14,15);
        }
        // B 32x16: lane=N (per half-wave), lanes 0-15 hold K{0..7,16..23}, 16-31 hold K{8..15,24..31}
        v16h bfrag[4];
#pragma unroll
        for (int tn = 0; tn < 4; ++tn) {
            const int br = wc * 64 + tn * 16 + lh;
            v8h d0 = *reinterpret_cast<const v8h*>(&Bs[cur][br * LDSK + (hi ? 8  : 0)]);
            v8h d1 = *reinterpret_cast<const v8h*>(&Bs[cur][br * LDSK + (hi ? 24 : 16)]);
            bfrag[tn] = __builtin_shufflevector(d0, d1,
                0,1,2,3,4,5,6,7,8,9,10,11,12,13,14,15);
        }

        // ---- 8 matrix ops per wave per K-step ----
#pragma unroll
        for (int tm = 0; tm < 2; ++tm)
#pragma unroll
            for (int tn = 0; tn < 4; ++tn)
                acc[tm][tn] = __builtin_amdgcn_wmma_f32_16x16x32_f16(
                    false, afrag[tm], false, bfrag[tn],
                    (short)0, acc[tm][tn], false, false);

        // next panel must have landed before anyone reads it
        if (tid < 32) __builtin_amdgcn_s_wait_tensorcnt(0);
        __syncthreads();
        cur ^= 1;
    }

    // ---- epilogue: C/D layout -> VGPR r: row = r + (hi?8:0), col = lh ----
#pragma unroll
    for (int tm = 0; tm < 2; ++tm) {
        const int rbase = i0 + wr * 32 + tm * 16 + (hi ? 8 : 0);
#pragma unroll
        for (int tn = 0; tn < 4; ++tn) {
            const int col = j0 + wc * 64 + tn * 16 + lh;
            if (col < N) {
#pragma unroll
                for (int r = 0; r < 8; ++r) {
                    const int row = rbase + r;
                    if (row < M) {
                        const float v = acc[tm][tn][r];
                        if (EPI == 0) {
                            reinterpret_cast<float*>(C)[(size_t)row * ldc + col] = v;
                        } else {
                            const float t = tanhf(ALPHA * v);
                            reinterpret_cast<_Float16*>(C)[(size_t)row * ldc + col] =
                                (_Float16)(t > 0.0f ? t : 0.0f);
                        }
                    }
                }
            }
        }
    }
}

// ---------------- top-32 per row: exact radix select + tie mask -------------
// adj >= 0 everywhere (H >= 0), so float ordering == uint-bit ordering.
__global__ __launch_bounds__(256) void topk_mask_kernel(float* __restrict__ adj, int n)
{
    __shared__ float    rowbuf[NNODES];   // 40 KB of the 320 KB WGP LDS
    __shared__ unsigned hist[256];
    __shared__ unsigned eqc[256];
    __shared__ unsigned eqx[256];
    __shared__ unsigned s_prefix, s_need;

    const int tid = threadIdx.x;
    const size_t base = (size_t)blockIdx.x * n;

    for (int j = tid; j < n; j += 256) rowbuf[j] = adj[base + j];
    if (tid == 0) { s_prefix = 0u; s_need = KSEL; }
    __syncthreads();

    unsigned prefix = 0u, need = KSEL;
#pragma unroll
    for (int pass = 0; pass < 4; ++pass) {
        const int shift = 24 - pass * 8;
        hist[tid] = 0u;
        __syncthreads();
        for (int j = tid; j < n; j += 256) {
            const unsigned u  = __float_as_uint(rowbuf[j]);
            const unsigned hb = (pass == 0) ? 0u : (u >> (shift + 8));
            if (hb == prefix) atomicAdd(&hist[(u >> shift) & 255u], 1u);
        }
        __syncthreads();
        if (tid == 0) {
            unsigned cum = 0u, nd = s_need, sel = 0u;
            for (int d = 255; d >= 0; --d) {
                const unsigned c = hist[d];
                if (cum + c >= nd) { sel = (unsigned)d; nd -= cum; break; }
                cum += c;
            }
            s_prefix = (s_prefix << 8) | sel;   // cnt(> vK) removed from need
            s_need   = nd;                      // # of ==vK still to keep
        }
        __syncthreads();
        prefix = s_prefix; need = s_need;
        __syncthreads();
    }

    // jax top_k tie-break = lowest index first: contiguous chunks + exclusive scan
    const int chunk = (n + 255) / 256;
    const int s = tid * chunk;
    int e = s + chunk; if (e > n) e = n;
    unsigned cnt = 0u;
    for (int j = s; j < e; ++j)
        if (__float_as_uint(rowbuf[j]) == prefix) ++cnt;
    eqc[tid] = cnt;
    __syncthreads();
    if (tid == 0) {
        unsigned run = 0u;
        for (int t = 0; t < 256; ++t) { eqx[t] = run; run += eqc[t]; }
    }
    __syncthreads();

    unsigned rank = eqx[tid];
    for (int j = s; j < e; ++j) {
        const float v = rowbuf[j];
        const unsigned u = __float_as_uint(v);
        bool keep = (u > prefix);
        if (u == prefix) { keep = (rank < need); ++rank; }
        adj[base + j] = keep ? v : 0.0f;
    }
}

// ------------------------------- launcher -----------------------------------
extern "C" void kernel_launch(void* const* d_in, const int* in_sizes, int n_in,
                              void* d_out, int out_size, void* d_ws, size_t ws_size,
                              hipStream_t stream)
{
    const int*   idx   = (const int*)  d_in[0];
    const float* embn  = (const float*)d_in[1];
    const float* embhe = (const float*)d_in[2];
    const float* W1    = (const float*)d_in[3];
    const float* b1    = (const float*)d_in[4];
    const float* W2    = (const float*)d_in[5];
    const float* b2    = (const float*)d_in[6];
    float* out = (float*)d_out;

    // workspace: ~12.9 MB of f16 activations
    _Float16* nv1 = (_Float16*)d_ws;                       // 10000*128
    _Float16* nv2 = nv1 + (size_t)NNODES * DIM;            //   512*128
    _Float16* H   = nv2 + (size_t)NHEDGES * DIM;           // 10000*512

    embed_mlp_kernel<<<NHEDGES, DIM, 0, stream>>>(embhe, W2, b2, nullptr, nv2, NHEDGES);
    embed_mlp_kernel<<<NNODES,  DIM, 0, stream>>>(embn,  W1, b1, idx,     nv1, NNODES);

    dim3 gH((NHEDGES + 127) / 128, (NNODES + 127) / 128);
    gemm_nt_wmma<1><<<gH, 256, 0, stream>>>(nv1, nv2, (void*)H,
                                            NNODES, NHEDGES, DIM, NHEDGES);

    dim3 gA((NNODES + 127) / 128, (NNODES + 127) / 128);
    gemm_nt_wmma<0><<<gA, 256, 0, stream>>>(H, H, (void*)out,
                                            NNODES, NNODES, NHEDGES, NNODES);

    topk_mask_kernel<<<NNODES, 256, 0, stream>>>(out, NNODES);
}
